// IBGuidedMultiHeadAttention_75694503625091
// MI455X (gfx1250) — matrix operand order repaired and checked by reading, not compiled, not run on hardware
//
#include <hip/hip_runtime.h>
#include <hip/hip_bf16.h>
#include <stdint.h>

typedef __attribute__((ext_vector_type(16))) _Float16 v16h;
typedef __attribute__((ext_vector_type(8)))  float    v8f;

#define BATCH  4
#define SEQ    2048
#define DMODEL 512
#define NHEAD  8
#define HDIM   64
#define MROWS  (BATCH*SEQ)   /* 8192 */

__device__ __forceinline__ v8f wmma_f16(v16h a, v16h b, v8f c) {
  // D = A(16x32 f16) * B(32x16 f16) + C(16x16 f32)
  return __builtin_amdgcn_wmma_f32_16x16x32_f16(false, a, false, b, (short)0, c, false, false);
}

__device__ __forceinline__ v8f zero8() {
  v8f z;
#pragma unroll
  for (int e = 0; e < 8; ++e) z[e] = 0.0f;
  return z;
}

// A-operand 16x32 f16 tile. Layout used here: lane holds row m=lane&15; half=lane>>4.
// 16 consecutive k starting at half*16 (matches load_b).
// src points at tile origin (row 0, k 0) of a row-major matrix, ld = row stride (halves).
__device__ __forceinline__ v16h load_a(const _Float16* src, int ld, int lane) {
  int m = lane & 15, half = lane >> 4;
  const _Float16* p = src + (size_t)m * ld;
  union { v16h h; uint4 u[2]; } r;
  r.u[0] = *reinterpret_cast<const uint4*>(p + half * 16);
  r.u[1] = *reinterpret_cast<const uint4*>(p + half * 16 + 8);
  return r.h;
}

// B-operand 32x16 f16 tile. Lane holds column n=lane&15 of B; 16 consecutive k
// starting at (lane>>4)*16. src = origin of row-major [N,K] (i.e. B transposed),
// so per-lane loads are contiguous.
__device__ __forceinline__ v16h load_b(const _Float16* src, int ld, int lane) {
  int n = lane & 15, grp = lane >> 4;
  const _Float16* p = src + (size_t)n * ld + grp * 16;
  union { v16h h; uint4 u[2]; } r;
  r.u[0] = *reinterpret_cast<const uint4*>(p);
  r.u[1] = *reinterpret_cast<const uint4*>(p + 8);
  return r.h;
}

// ---------------------------------------------------------------- conversions
__global__ __launch_bounds__(256) void cvt_f32_f16(const float* __restrict__ src,
                                                   _Float16* __restrict__ dst, int n) {
  for (int i = blockIdx.x * blockDim.x + threadIdx.x; i < n; i += gridDim.x * blockDim.x)
    dst[i] = (_Float16)src[i];
}

// ------------------------------------------------- confidence stats (per key token)
__global__ __launch_bounds__(256) void conf_stats(const float* __restrict__ kc,
                                                  float* __restrict__ cpos,
                                                  float* __restrict__ cbias) {
  int lane = threadIdx.x & 31, wid = threadIdx.x >> 5;
  int row = blockIdx.x * 8 + wid;                 // 0..8191
  const float* p = kc + (size_t)row * DMODEL;
  float s = 0.0f;
#pragma unroll
  for (int i = 0; i < DMODEL / 32; ++i) s += p[lane + i * 32];
#pragma unroll
  for (int m = 16; m >= 1; m >>= 1) s += __shfl_xor(s, m, 32);
  if (lane == 0) {
    float mean = s * (1.0f / DMODEL);
    cpos[row]  = mean;
    cbias[row] = __logf(fmaxf(mean, 1e-6f));
  }
}

// ---------------------------------------------------------------- QKV projection
// out = X @ W^T + b ; Q,K -> [B,H,T,HD] f16 ; V -> [B,H,HD,T] f16, scaled by conf_pos.
__global__ __launch_bounds__(256) void gemm_qkv(
    const _Float16* __restrict__ Xq, const _Float16* __restrict__ Xk, const _Float16* __restrict__ Xv,
    const _Float16* __restrict__ Wq, const _Float16* __restrict__ Wk, const _Float16* __restrict__ Wv,
    const float* __restrict__ bq, const float* __restrict__ bk, const float* __restrict__ bv,
    const float* __restrict__ cpos,
    _Float16* __restrict__ Qh, _Float16* __restrict__ Kh, _Float16* __restrict__ Vt) {
  int lane = threadIdx.x & 31, wid = threadIdx.x >> 5;
  int mt = blockIdx.x * 8 + wid;            // 0..511 : 16-row tile of [8192,512]
  int n0 = blockIdx.y * 64;                 // 4 output tiles of 16
  int z  = blockIdx.z;                      // 0=Q 1=K 2=V (uniform)
  const _Float16* X    = (z == 0) ? Xq : (z == 1) ? Xk : Xv;
  const _Float16* W    = (z == 0) ? Wq : (z == 1) ? Wk : Wv;
  const float*    bias = (z == 0) ? bq : (z == 1) ? bk : bv;

  v8f acc[4];
#pragma unroll
  for (int t = 0; t < 4; ++t) acc[t] = zero8();

  const _Float16* Xrow = X + (size_t)mt * 16 * DMODEL;
  for (int kb = 0; kb < DMODEL; kb += 32) {
    v16h a = load_a(Xrow + kb, DMODEL, lane);
#pragma unroll
    for (int t = 0; t < 4; ++t) {
      v16h b = load_b(W + (size_t)(n0 + t * 16) * DMODEL + kb, DMODEL, lane);
      acc[t] = wmma_f16(a, b, acc[t]);
    }
  }

  int nl = lane & 15, hf = lane >> 4;
#pragma unroll
  for (int t = 0; t < 4; ++t) {
    int n = n0 + t * 16 + nl;
    float bn = bias[n];
    int h = n >> 6, hd = n & 63;
#pragma unroll
    for (int r = 0; r < 8; ++r) {
      int m = mt * 16 + r + 8 * hf;         // global token row
      int b = m >> 11, tok = m & 2047;
      float v = acc[t][r] + bn;
      if (z < 2) {
        _Float16* dst = (z == 0) ? Qh : Kh;
        dst[(((size_t)(b * NHEAD + h)) * SEQ + tok) * HDIM + hd] = (_Float16)v;
      } else {
        v *= cpos[m];
        Vt[(((size_t)(b * NHEAD + h)) * HDIM + hd) * SEQ + tok] = (_Float16)v;
      }
    }
  }
}

// ---------------------------------------------------------------- flash attention
__global__ __launch_bounds__(256) void attn_kernel(
    const _Float16* __restrict__ Qh, const _Float16* __restrict__ Kh,
    const _Float16* __restrict__ Vt, const float* __restrict__ cbias,
    const unsigned char* __restrict__ kmask, const float* __restrict__ conf_scale_p,
    _Float16* __restrict__ ctx) {
  __shared__ _Float16 sP[8][16 * 40];       // per-wave 16x32 P tile, stride 40 halves
  int lane = threadIdx.x & 31, wid = threadIdx.x >> 5;
  int bh = blockIdx.y; int b = bh >> 3, h = bh & 7;
  int qbase = (blockIdx.x * 8 + wid) * 16;
  const float cs = *conf_scale_p;

  const _Float16* Qbh = Qh + ((size_t)bh * SEQ) * HDIM;
  const _Float16* Kbh = Kh + ((size_t)bh * SEQ) * HDIM;
  const _Float16* Vbh = Vt + ((size_t)bh * HDIM) * SEQ;
  const float* cb = cbias + (size_t)b * SEQ;
  const unsigned char* mk = kmask + (size_t)b * SEQ;

  v16h qa0 = load_a(Qbh + (size_t)qbase * HDIM, HDIM, lane);
  v16h qa1 = load_a(Qbh + (size_t)qbase * HDIM + 32, HDIM, lane);

  float mrun[8], lrun[8];
  v8f o[4];
#pragma unroll
  for (int r = 0; r < 8; ++r) { mrun[r] = -3.0e38f; lrun[r] = 0.0f; }
#pragma unroll
  for (int t = 0; t < 4; ++t) o[t] = zero8();

  _Float16* myP = &sP[wid][0];
  int nl = lane & 15, hf = lane >> 4;

  for (int kb = 0; kb < SEQ; kb += 32) {
    if (kb + 32 < SEQ)                       // uniform branch: prefetch next K tile
      __builtin_prefetch(Kbh + (size_t)(kb + 32) * HDIM, 0, 3);

    v8f s0 = zero8(), s1 = zero8();
    {
      v16h k0 = load_b(Kbh + (size_t)kb * HDIM, HDIM, lane);
      v16h k1 = load_b(Kbh + (size_t)(kb + 16) * HDIM, HDIM, lane);
      s0 = wmma_f16(qa0, k0, s0);
      s1 = wmma_f16(qa0, k1, s1);
      v16h k2 = load_b(Kbh + (size_t)kb * HDIM + 32, HDIM, lane);
      v16h k3 = load_b(Kbh + (size_t)(kb + 16) * HDIM + 32, HDIM, lane);
      s0 = wmma_f16(qa1, k2, s0);
      s1 = wmma_f16(qa1, k3, s1);
    }

    float bias0 = cs * cb[kb + nl]      + (mk[kb + nl]      ? 0.0f : -3.0e38f);
    float bias1 = cs * cb[kb + 16 + nl] + (mk[kb + 16 + nl] ? 0.0f : -3.0e38f);

    float p0[8], p1[8], alpha[8];
#pragma unroll
    for (int r = 0; r < 8; ++r) {
      float v0 = s0[r] * 0.125f + bias0;    // 1/sqrt(64)
      float v1 = s1[r] * 0.125f + bias1;
      float rm = fmaxf(v0, v1);
#pragma unroll
      for (int x = 1; x < 16; x <<= 1) rm = fmaxf(rm, __shfl_xor(rm, x, 32));
      float nm = fmaxf(mrun[r], rm);
      alpha[r] = __expf(mrun[r] - nm);
      mrun[r]  = nm;
      p0[r] = __expf(v0 - nm);
      p1[r] = __expf(v1 - nm);
      float rs = p0[r] + p1[r];
#pragma unroll
      for (int x = 1; x < 16; x <<= 1) rs += __shfl_xor(rs, x, 32);
      lrun[r] = lrun[r] * alpha[r] + rs;
    }

#pragma unroll
    for (int r = 0; r < 8; ++r) {
      int mrow = r + 8 * hf;
      myP[mrow * 40 + nl]      = (_Float16)p0[r];
      myP[mrow * 40 + 16 + nl] = (_Float16)p1[r];
#pragma unroll
      for (int t = 0; t < 4; ++t) o[t][r] *= alpha[r];
    }

    __asm__ volatile("s_wait_dscnt 0" ::: "memory");   // cross-lane P visibility
    v16h pa = load_a(myP, 40, lane);

#pragma unroll
    for (int t = 0; t < 4; ++t) {
      v16h vb = load_b(Vbh + (size_t)(t * 16) * SEQ + kb, SEQ, lane);
      o[t] = wmma_f16(pa, vb, o[t]);
    }
  }

#pragma unroll
  for (int t = 0; t < 4; ++t) {
#pragma unroll
    for (int r = 0; r < 8; ++r) {
      int tok = qbase + r + 8 * hf;
      float v = o[t][r] / lrun[r];
      ctx[((size_t)(b * SEQ + tok)) * DMODEL + h * HDIM + t * 16 + nl] = (_Float16)v;
    }
  }
}

// ---------------------------------------------------------------- output projection
__global__ __launch_bounds__(256) void gemm_out(
    const _Float16* __restrict__ Xc, const _Float16* __restrict__ Wo,
    const float* __restrict__ bo, float* __restrict__ out) {
  int lane = threadIdx.x & 31, wid = threadIdx.x >> 5;
  int mt = blockIdx.x * 8 + wid;
  int n0 = blockIdx.y * 64;

  v8f acc[4];
#pragma unroll
  for (int t = 0; t < 4; ++t) acc[t] = zero8();

  const _Float16* Xrow = Xc + (size_t)mt * 16 * DMODEL;
  for (int kb = 0; kb < DMODEL; kb += 32) {
    v16h a = load_a(Xrow + kb, DMODEL, lane);
#pragma unroll
    for (int t = 0; t < 4; ++t) {
      v16h b = load_b(Wo + (size_t)(n0 + t * 16) * DMODEL + kb, DMODEL, lane);
      acc[t] = wmma_f16(a, b, acc[t]);
    }
  }

  int nl = lane & 15, hf = lane >> 4;
#pragma unroll
  for (int t = 0; t < 4; ++t) {
    int n = n0 + t * 16 + nl;
    float bn = bo[n];
#pragma unroll
    for (int r = 0; r < 8; ++r) {
      int m = mt * 16 + r + 8 * hf;
      out[(size_t)m * DMODEL + n] = acc[t][r] + bn;
    }
  }
}

// ---------------------------------------------------------------------- launcher
extern "C" void kernel_launch(void* const* d_in, const int* in_sizes, int n_in,
                              void* d_out, int out_size, void* d_ws, size_t ws_size,
                              hipStream_t stream) {
  const float* query = (const float*)d_in[0];
  const float* key   = (const float*)d_in[1];
  const float* value = (const float*)d_in[2];
  const float* kconf = (const float*)d_in[3];
  const unsigned char* kmask = (const unsigned char*)d_in[4];
  const float* Wq = (const float*)d_in[5];
  const float* bq = (const float*)d_in[6];
  const float* Wk = (const float*)d_in[7];
  const float* bk = (const float*)d_in[8];
  const float* Wv = (const float*)d_in[9];
  const float* bv = (const float*)d_in[10];
  const float* Wo = (const float*)d_in[11];
  const float* bo = (const float*)d_in[12];
  const float* conf_scale = (const float*)d_in[13];
  float* out = (float*)d_out;

  char* ws = (char*)d_ws;
  size_t off = 0;
  auto alloc = [&](size_t bytes) {
    char* p = ws + off;
    off = (off + bytes + 255) & ~(size_t)255;
    return p;
  };
  const size_t actB = (size_t)MROWS * DMODEL * sizeof(_Float16);   // 8 MiB
  const size_t wB   = (size_t)DMODEL * DMODEL * sizeof(_Float16);  // 0.5 MiB
  _Float16* Xqh = (_Float16*)alloc(actB);
  _Float16* Xkh = (_Float16*)alloc(actB);
  _Float16* Xvh = (_Float16*)alloc(actB);
  _Float16* Wqh = (_Float16*)alloc(wB);
  _Float16* Wkh = (_Float16*)alloc(wB);
  _Float16* Wvh = (_Float16*)alloc(wB);
  _Float16* Woh = (_Float16*)alloc(wB);
  _Float16* Qh  = (_Float16*)alloc(actB);
  _Float16* Kh  = (_Float16*)alloc(actB);
  _Float16* Vt  = (_Float16*)alloc(actB);
  _Float16* Ctx = (_Float16*)alloc(actB);
  float* cpos  = (float*)alloc((size_t)MROWS * sizeof(float));
  float* cbias = (float*)alloc((size_t)MROWS * sizeof(float));
  (void)in_sizes; (void)n_in; (void)out_size; (void)ws_size;

  cvt_f32_f16<<<1024, 256, 0, stream>>>(query, Xqh, MROWS * DMODEL);
  cvt_f32_f16<<<1024, 256, 0, stream>>>(key,   Xkh, MROWS * DMODEL);
  cvt_f32_f16<<<1024, 256, 0, stream>>>(value, Xvh, MROWS * DMODEL);
  cvt_f32_f16<<<512, 256, 0, stream>>>(Wq, Wqh, DMODEL * DMODEL);
  cvt_f32_f16<<<512, 256, 0, stream>>>(Wk, Wkh, DMODEL * DMODEL);
  cvt_f32_f16<<<512, 256, 0, stream>>>(Wv, Wvh, DMODEL * DMODEL);
  cvt_f32_f16<<<512, 256, 0, stream>>>(Wo, Woh, DMODEL * DMODEL);

  conf_stats<<<MROWS / 8, 256, 0, stream>>>(kconf, cpos, cbias);

  gemm_qkv<<<dim3(MROWS / 128, DMODEL / 64, 3), 256, 0, stream>>>(
      Xqh, Xkh, Xvh, Wqh, Wkh, Wvh, bq, bk, bv, cpos, Qh, Kh, Vt);

  attn_kernel<<<dim3(SEQ / 128, BATCH * NHEAD), 256, 0, stream>>>(
      Qh, Kh, Vt, cbias, kmask, conf_scale, Ctx);

  gemm_out<<<dim3(MROWS / 128, DMODEL / 64), 256, 0, stream>>>(Ctx, Woh, bo, out);
}
